// FeatureExtaction_26895085207605
// MI455X (gfx1250) — compile-verified, matrix operand order checked by reading, not run on hardware
//
#include <hip/hip_runtime.h>
#include <hip/hip_bf16.h>

// ---------------- problem dims ----------------
#define B_  128
#define T_  512
#define E_  300
#define H_  150
#define KE  320          // E padded to mult of 32 (WMMA K)
#define KH  160          // H padded to mult of 32
#define HP  160          // hidden padded per gate
#define NG  640          // 4 * HP gate columns
#define NT_ 40           // NG/16 N-tiles
#define BT  (B_ * T_)

typedef __attribute__((ext_vector_type(16))) _Float16 v16h;
typedef __attribute__((ext_vector_type(8)))  _Float16 v8h;
typedef __attribute__((ext_vector_type(8)))  float    v8f;

union Frag16 { v16h h; uint4 q[2]; };
union Frag8  { v8h  h; uint4 q;    };

__device__ __forceinline__ float sigmoidf_(float x) {
    return 1.0f / (1.0f + __expf(-x));
}

// ---------------- workspace layout (bytes) ----------------
constexpr size_t OFF_XH   = 0;
constexpr size_t SZ_XH    = (size_t)BT * KE * 2;        // x transposed to [t*B+b][k], f16
constexpr size_t OFF_WIH  = OFF_XH + SZ_XH;
constexpr size_t SZ_WIH   = (size_t)2 * NG * KE * 2;    // W_ih padded, [d][n][k] f16
constexpr size_t OFF_WHH  = OFF_WIH + SZ_WIH;
constexpr size_t SZ_WHH   = (size_t)2 * NG * KH * 2;    // W_hh padded, [d][n][k] f16
constexpr size_t OFF_BP   = OFF_WHH + SZ_WHH;
constexpr size_t SZ_BP    = (size_t)2 * NG * 4;         // bias padded f32
constexpr size_t OFF_XG   = OFF_BP + SZ_BP;
// xg stored fragment-native: [d][T][btile 8][ntile 40][lane 32][8 halves]
constexpr size_t SZ_XG    = (size_t)2 * T_ * 8 * NT_ * 32 * 8 * 2;
constexpr size_t OFF_MEAN = OFF_XG + SZ_XG;
constexpr size_t SZ_MEAN  = (size_t)B_ * 2 * HP * 4;
constexpr size_t OFF_HK   = OFF_MEAN + SZ_MEAN;

// ---------------- prep kernels ----------------
// transpose x[b][t][e] -> xh[(t*B + b)][k] (f16, K-padded) so GEMM M-tiles are
// 16 batch rows at a fixed timestep (matches the scan's per-step tile).
__global__ void prep_x_kernel(const float* __restrict__ x, _Float16* __restrict__ xh) {
    size_t i = (size_t)blockIdx.x * blockDim.x + threadIdx.x;   // over BT*KE
    if (i >= (size_t)BT * KE) return;
    int    k   = (int)(i % KE);
    size_t row = i / KE;                 // = t*B + b
    int t = (int)(row / B_);
    int b = (int)(row % B_);
    xh[i] = (k < E_) ? (_Float16)x[((size_t)b * T_ + t) * E_ + k] : (_Float16)0.0f;
}

__global__ void prep_w_kernel(const float* __restrict__ wf, const float* __restrict__ wb,
                              _Float16* __restrict__ wp, int Ksrc, int Kp) {
    size_t i = (size_t)blockIdx.x * blockDim.x + threadIdx.x;   // over 2*NG*Kp
    if (i >= (size_t)2 * NG * Kp) return;
    int k = (int)(i % Kp);
    int n = (int)((i / Kp) % NG);
    int d = (int)(i / ((size_t)Kp * NG));
    int g = n / HP, j = n % HP;
    const float* w = d ? wb : wf;
    float v = (j < H_ && k < Ksrc) ? w[(size_t)(g * H_ + j) * Ksrc + k] : 0.0f;
    wp[i] = (_Float16)v;
}

__global__ void prep_b_kernel(const float* __restrict__ bf, const float* __restrict__ bb,
                              float* __restrict__ bp) {
    int i = blockIdx.x * blockDim.x + threadIdx.x;              // over 2*NG
    if (i >= 2 * NG) return;
    int n = i % NG, d = i / NG;
    int g = n / HP, j = n % HP;
    const float* b = d ? bb : bf;
    bp[i] = (j < H_) ? b[g * H_ + j] : 0.0f;
}

// ---------------- input-projection GEMM: xg = xh @ W_ih^T + b ----------------
// grid: (T, NG/64, 2), block 256 (8 waves). blockIdx.x = timestep, wave = batch tile.
// B-panel (64 x 320 f16 = 40KB) in LDS shared by the 8 waves; A double-buffered
// from global; B fragments double-buffered from LDS; D stored fragment-native.
__global__ void gemm_xg_kernel(const _Float16* __restrict__ xh,
                               const _Float16* __restrict__ wihP,
                               const float*    __restrict__ bpad,
                               _Float16*       __restrict__ xg) {
    __shared__ __align__(16) _Float16 bpanel[64 * KE];          // 40 KB

    const int tid  = threadIdx.x;
    const int lane = tid & 31;
    const int wv   = tid >> 5;                 // batch tile 0..7
    const int t    = blockIdx.x;               // timestep
    const int nb   = blockIdx.y * 64;          // N base
    const int d    = blockIdx.z;
    const int lm   = lane & 15;
    const int koff = (lane < 16) ? 0 : 8;      // ISA 16-bit A/B K placement

    // stage the block's B panel (rows nb..nb+63 of wihP, contiguous) into LDS
    {
        const uint4* src = (const uint4*)(wihP + ((size_t)d * NG + nb) * KE);
        uint4*       dst = (uint4*)bpanel;
        for (int i = tid; i < 64 * KE / 8; i += blockDim.x) dst[i] = src[i];
    }
    __syncthreads();

    const int mt = t * 8 + wv;                 // M tile over rows (t*B + b)
    const uint4* Aq = (const uint4*)(xh + (size_t)(mt * 16 + lm) * KE);
    const uint4* Bq[4];
#pragma unroll
    for (int nt = 0; nt < 4; ++nt)
        Bq[nt] = (const uint4*)(bpanel + (size_t)(nt * 16 + lm) * KE);

    v8f acc[4] = {};
    Frag16 a[2], bf[2];
    // prologue: chunk-0 A fragment and (kk=0,nt=0) B fragment
    a[0].q[0]  = Aq[(0 + koff) >> 3];
    a[0].q[1]  = Aq[(16 + koff) >> 3];
    bf[0].q[0] = Bq[0][koff >> 3];
    bf[0].q[1] = Bq[0][(koff + 16) >> 3];

#pragma unroll
    for (int kk = 0; kk < KE / 32; ++kk) {
        const int curA = kk & 1;
        if (kk + 1 < KE / 32) {                // double-buffer next A chunk (global)
            const int k0n = (kk + 1) * 32 + koff;
            a[curA ^ 1].q[0] = Aq[k0n >> 3];
            a[curA ^ 1].q[1] = Aq[(k0n + 16) >> 3];
        }
#pragma unroll
        for (int nt = 0; nt < 4; ++nt) {
            const int i   = kk * 4 + nt;       // flattened (kk,nt)
            const int cur = i & 1;
            if (i + 1 < (KE / 32) * 4) {       // double-buffer next B frag (LDS)
                const int kkn = (i + 1) >> 2;
                const int ntn = (i + 1) & 3;
                const int k0n = kkn * 32 + koff;
                bf[cur ^ 1].q[0] = Bq[ntn][k0n >> 3];
                bf[cur ^ 1].q[1] = Bq[ntn][(k0n + 16) >> 3];
            }
            acc[nt] = __builtin_amdgcn_wmma_f32_16x16x32_f16(
                false, a[curA].h, false, bf[cur].h, (short)0, acc[nt], false, false);
        }
    }

    // fragment-native store: one contiguous 16B store per lane per N-tile
    const float* bp = bpad + d * NG;
    uint4* xq = (uint4*)xg;
#pragma unroll
    for (int nt = 0; nt < 4; ++nt) {
        const int ntile = blockIdx.y * 4 + nt;
        const float bias = bp[ntile * 16 + lm];
        Frag8 o;
#pragma unroll
        for (int v = 0; v < 8; ++v) o.h[v] = (_Float16)(acc[nt][v] + bias);
        const size_t qidx = ((((size_t)d * T_ + t) * 8 + wv) * NT_ + ntile) * 32 + lane;
        xq[qidx] = o.q;
    }
}

// ---------------- fused LSTM scan + pooling ----------------
// grid: (B/16, 2 dirs), block 320 (10 waves). Wave wv owns hidden cols [wv*16,+16)
// for all 4 gates. W_hh (200KB) + double-buffered h (2x5KB) in LDS.
// xg C-operands: 4 coalesced b128 loads/lane pipelined one step ahead; W_hh B
// fragments pipelined one WMMA ahead inside the step.
__global__ void lstm_scan_kernel(const _Float16* __restrict__ xg,
                                 const _Float16* __restrict__ whhP,
                                 const int*      __restrict__ lens,
                                 float* __restrict__ mean_ws,
                                 float* __restrict__ hk_ws) {
    extern __shared__ __align__(16) char smem[];
    _Float16* whh_s = (_Float16*)smem;                            // NG*KH halves
    _Float16* h_s0  = (_Float16*)(smem + (size_t)NG * KH * 2);    // 16*KH halves
    _Float16* h_s1  = h_s0 + 16 * KH;

    const int bx   = blockIdx.x;               // batch tile
    const int b0   = bx * 16;
    const int d    = blockIdx.y;
    const int tid  = threadIdx.x;
    const int lane = tid & 31;
    const int wv   = tid >> 5;                 // j-tile 0..9
    const int lm   = lane & 15;
    const int koff = (lane < 16) ? 0 : 8;
    const int hi8  = (lane < 16) ? 0 : 8;

    // stage W_hh into LDS, zero both h buffers
    {
        const uint4* src = (const uint4*)(whhP + (size_t)d * NG * KH);
        uint4*       dst = (uint4*)whh_s;
        for (int i = tid; i < NG * KH / 8; i += blockDim.x) dst[i] = src[i];
        uint4 z; z.x = z.y = z.z = z.w = 0u;
        uint4* hz = (uint4*)h_s0;
        for (int i = tid; i < 2 * 16 * KH / 8; i += blockDim.x) hz[i] = z;
    }
    __syncthreads();

    int Lv[8];
#pragma unroll
    for (int v = 0; v < 8; ++v) Lv[v] = lens[b0 + v + hi8];

    const uint4* Bq[4];
#pragma unroll
    for (int g = 0; g < 4; ++g)
        Bq[g] = (const uint4*)(whh_s + (size_t)(g * HP + wv * 16 + lm) * KH);

    float c_r[8] = {}, msum[8] = {}, hk[8] = {};
    const uint4* xq = (const uint4*)xg;

    // prologue: fetch step-0 xg fragments
    uint4 cq[4];
    {
        const int tt0 = (d == 0) ? 0 : (T_ - 1);
#pragma unroll
        for (int g = 0; g < 4; ++g)
            cq[g] = xq[((((size_t)d * T_ + tt0) * 8 + bx) * NT_ + (g * 10 + wv)) * 32 + lane];
    }

    for (int t = 0; t < T_; ++t) {
        const int tt = (d == 0) ? t : (T_ - 1 - t);   // original time index

        // C from the pipelined xg fragments
        v8f acc[4];
#pragma unroll
        for (int g = 0; g < 4; ++g) {
            Frag8 u; u.q = cq[g];
#pragma unroll
            for (int v = 0; v < 8; ++v) acc[g][v] = (float)u.h[v];
        }
        // issue next step's xg fragment loads now; latency hides under WMMAs
        if (t + 1 < T_) {
            const int tn = (d == 0) ? (t + 1) : (T_ - 2 - t);
#pragma unroll
            for (int g = 0; g < 4; ++g)
                cq[g] = xq[((((size_t)d * T_ + tn) * 8 + bx) * NT_ + (g * 10 + wv)) * 32 + lane];
        }

        // A fragments of h from current LDS buffer (K = 160 -> 5 chunks of 32)
        _Float16* h_rd = (t & 1) ? h_s1 : h_s0;
        _Float16* h_wr = (t & 1) ? h_s0 : h_s1;
        Frag16 a[5];
        {
            const uint4* Ahq = (const uint4*)(h_rd + lm * KH);
#pragma unroll
            for (int kk = 0; kk < 5; ++kk) {
                const int k0 = kk * 32 + koff;
                a[kk].q[0] = Ahq[k0 >> 3];
                a[kk].q[1] = Ahq[(k0 + 16) >> 3];
            }
        }

        // 20 WMMAs on h @ W_hh^T with B fragments pipelined one WMMA ahead
        Frag16 bb[2];
        bb[0].q[0] = Bq[0][koff >> 3];
        bb[0].q[1] = Bq[0][(koff + 16) >> 3];
#pragma unroll
        for (int i = 0; i < 20; ++i) {         // i = g*5 + kk
            const int cur = i & 1;
            if (i + 1 < 20) {
                const int gn  = (i + 1) / 5;
                const int k0n = ((i + 1) % 5) * 32 + koff;
                bb[cur ^ 1].q[0] = Bq[gn][k0n >> 3];
                bb[cur ^ 1].q[1] = Bq[gn][(k0n + 16) >> 3];
            }
            const int g  = i / 5;
            const int kk = i % 5;
            acc[g] = __builtin_amdgcn_wmma_f32_16x16x32_f16(
                false, a[kk].h, false, bb[cur].h, (short)0, acc[g], false, false);
        }

        // gate nonlinearities, cell/hidden update, fused pooling
        const int ncol = wv * 16 + lm;
#pragma unroll
        for (int v = 0; v < 8; ++v) {
            const float ig = sigmoidf_(acc[0][v]);
            const float fg = sigmoidf_(acc[1][v]);
            const float gg = tanhf(acc[2][v]);
            const float og = sigmoidf_(acc[3][v]);
            const float c  = fg * c_r[v] + ig * gg;
            c_r[v] = c;
            const float h = og * tanhf(c);
            if (tt < Lv[v])      msum[v] += h;     // masked mean accumulator
            if (tt == Lv[v] - 1) hk[v] = h;        // last valid hidden state
            h_wr[(v + hi8) * KH + ncol] = (_Float16)h;
        }
        __syncthreads();   // single barrier: new h visible before next step reads it
    }

#pragma unroll
    for (int v = 0; v < 8; ++v) {
        const int b = b0 + v + hi8;
        const size_t idx = ((size_t)b * 2 + d) * HP + wv * 16 + lm;
        mean_ws[idx] = msum[v];
        hk_ws[idx]   = hk[v];
    }
}

// ---------------- finalize: out = tanh([mean/L , h_k]) ----------------
__global__ void finalize_kernel(const float* __restrict__ mean_ws,
                                const float* __restrict__ hk_ws,
                                const int*   __restrict__ lens,
                                float* __restrict__ out) {
    const int b = blockIdx.x;
    const int c = threadIdx.x;
    if (c >= 4 * H_) return;
    const int part = c / (2 * H_);       // 0: mean, 1: h_k
    const int hcol = c % (2 * H_);
    const int dd   = hcol / H_;
    const int j    = hcol % H_;
    const size_t idx = ((size_t)b * 2 + dd) * HP + j;
    float v = (part == 0) ? mean_ws[idx] / (float)lens[b] : hk_ws[idx];
    out[(size_t)b * 4 * H_ + c] = tanhf(v);
}

// ---------------- launcher ----------------
extern "C" void kernel_launch(void* const* d_in, const int* in_sizes, int n_in,
                              void* d_out, int out_size, void* d_ws, size_t ws_size,
                              hipStream_t stream) {
    const float* x      = (const float*)d_in[0];
    const int*   lens   = (const int*)  d_in[1];
    const float* W_ih_f = (const float*)d_in[2];
    const float* W_hh_f = (const float*)d_in[3];
    const float* b_f    = (const float*)d_in[4];
    const float* W_ih_b = (const float*)d_in[5];
    const float* W_hh_b = (const float*)d_in[6];
    const float* b_b    = (const float*)d_in[7];
    float* out = (float*)d_out;

    char* ws = (char*)d_ws;
    _Float16* xh    = (_Float16*)(ws + OFF_XH);
    _Float16* wihP  = (_Float16*)(ws + OFF_WIH);
    _Float16* whhP  = (_Float16*)(ws + OFF_WHH);
    float*    bpad  = (float*)   (ws + OFF_BP);
    _Float16* xg    = (_Float16*)(ws + OFF_XG);
    float*    meanw = (float*)   (ws + OFF_MEAN);
    float*    hkw   = (float*)   (ws + OFF_HK);

    // 1) pad/convert/transpose inputs to f16
    {
        size_t n = (size_t)BT * KE;
        prep_x_kernel<<<dim3((unsigned)((n + 255) / 256)), dim3(256), 0, stream>>>(x, xh);
    }
    {
        size_t n = (size_t)2 * NG * KE;
        prep_w_kernel<<<dim3((unsigned)((n + 255) / 256)), dim3(256), 0, stream>>>(
            W_ih_f, W_ih_b, wihP, E_, KE);
    }
    {
        size_t n = (size_t)2 * NG * KH;
        prep_w_kernel<<<dim3((unsigned)((n + 255) / 256)), dim3(256), 0, stream>>>(
            W_hh_f, W_hh_b, whhP, H_, KH);
    }
    prep_b_kernel<<<dim3(5), dim3(256), 0, stream>>>(b_f, b_b, bpad);

    // 2) input-projection GEMM (LDS B-panel, double-buffered A+B, fragment-native D)
    gemm_xg_kernel<<<dim3(T_, NG / 64, 2), dim3(256), 0, stream>>>(
        xh, wihP, bpad, xg);

    // 3) fused recurrent scan + pooling (W_hh resident in LDS, 1 barrier/step,
    //    xg fragments pipelined one step ahead, B frags one WMMA ahead)
    {
        const size_t shmem = (size_t)NG * KH * 2 + (size_t)2 * 16 * KH * 2; // 215,040 B
        lstm_scan_kernel<<<dim3(B_ / 16, 2), dim3(320), shmem, stream>>>(
            xg, whhP, lens, meanw, hkw);
    }

    // 4) finalize
    finalize_kernel<<<dim3(B_), dim3(640), 0, stream>>>(meanw, hkw, lens, out);
}
